// MambaBlockDP_26061861552217
// MI455X (gfx1250) — matrix operand order repaired and checked by reading, not compiled, not run on hardware
//
#include <hip/hip_runtime.h>

// ---------------- MambaBlock for MI455X (gfx1250, wave32, WMMA) ----------------
// B=2, L=1024, D_MODEL=1024, D_INNER=2048, DT_RANK=64, D_STATE=16, D_CONV=4
// GEMMs: bf16 WMMA (fp32 acc) with async Global->LDS DMA double buffering.
// Scan: fp32 VALU, register-resident state.

#define BATCH 2
#define SEQL 1024
#define DMODEL 1024
#define DINNER 2048
#define DTRANK 64
#define DSTATE 16
#define MTOK (BATCH * SEQL) // 2048 rows everywhere

typedef __bf16 bf16_t;
typedef __attribute__((ext_vector_type(16))) __bf16 v16bf;
typedef __attribute__((ext_vector_type(8))) float v8f;

union FragU {
    unsigned int u[8];
    v16bf v;
};

__device__ __forceinline__ unsigned short f2bf(float f) {
    unsigned int u = __float_as_uint(f);
    unsigned int r = (u + 0x7FFFu + ((u >> 16) & 1u)) >> 16;
    return (unsigned short)r;
}

__device__ __forceinline__ float silu_f(float x) {
    return x * (1.0f / (1.0f + __expf(-x)));
}

// CDNA5 async Global->LDS DMA (ASYNCcnt-tracked, no VGPR round trip).
// lds_off: wave-relative LDS byte offset (16B aligned); gaddr: 64-bit global.
__device__ __forceinline__ void async_ld_b128(unsigned lds_off,
                                              unsigned long long gaddr) {
    asm volatile("global_load_async_to_lds_b128 %0, %1, off"
                 :
                 : "v"(lds_off), "v"(gaddr)
                 : "memory");
}
__device__ __forceinline__ void wait_async3() {
    asm volatile("s_wait_asynccnt 0x3" ::: "memory");
}
__device__ __forceinline__ void wait_async0() {
    asm volatile("s_wait_asynccnt 0x0" ::: "memory");
}

// ---------------------------------------------------------------------------
// Generic tiled WMMA GEMM: C[M,N] = epilogue( A[M,K] @ W[N,K]^T )
// A, W are bf16 (stored as u16) row-major. flags: 1=+bias[c], 2=gelu,
// 4=softplus, 8=+resid[r*ldc+c], 16=store bf16 (else fp32).
// Block tile 128x64, K-step 32, 8 waves (4x2), 32x32 per wave, double-buffered
// LDS fed by global_load_async_to_lds_b128.
// ---------------------------------------------------------------------------
#define TM 128
#define TN 64
#define TK 32
#define LDSP 40 // padded LDS row pitch in bf16 elements (80B, 16B aligned)

__global__ __launch_bounds__(256) void gemm_bf16_wmma(
    const unsigned short* __restrict__ A, const unsigned short* __restrict__ W,
    void* __restrict__ Cout, const float* __restrict__ bias,
    const float* __restrict__ resid, int M, int N, int K, int lda, int ldb,
    int ldc, int flags) {
    __shared__ unsigned short As[2 * TM * LDSP];
    __shared__ unsigned short Bs[2 * TN * LDSP];
    const unsigned ASZ = TM * LDSP * 2; // bytes per A buffer
    const unsigned BSZ = TN * LDSP * 2; // bytes per B buffer

    const int t = threadIdx.x;
    const int lane = t & 31;
    const int w = t >> 5;        // 8 waves
    const int wm = w >> 1;       // 0..3  (32-row strips)
    const int wn = w & 1;        // 0..1  (32-col strips)
    const int m0 = blockIdx.x * TM;
    const int n0 = blockIdx.y * TN;
    const int khalf = lane >> 4; // 0 or 1
    const int l16 = lane & 15;

    // ---- async staging map: per thread 2 A-chunks + 1 B-chunk of 16B ----
    const int arow = t >> 2; // 0..63 (and +64 for second chunk)
    const int ac = t & 3;    // 16B chunk within 64B k-row
    const int brow = t >> 2; // 0..63
    const int bc = t & 3;
    int beff = n0 + brow;
    if (beff > N - 1) beff = N - 1; // clamp; padded cols masked in epilogue

    const unsigned asbase = (unsigned)(unsigned long long)(void*)&As[0];
    const unsigned bsbase = (unsigned)(unsigned long long)(void*)&Bs[0];
    const unsigned a0l = asbase + (unsigned)(arow * LDSP + ac * 8) * 2;
    const unsigned a1l = asbase + (unsigned)((64 + arow) * LDSP + ac * 8) * 2;
    const unsigned bl = bsbase + (unsigned)(brow * LDSP + bc * 8) * 2;

    unsigned long long Aaddr0 =
        (unsigned long long)(A + (size_t)(m0 + arow) * lda + ac * 8);
    unsigned long long Aaddr1 =
        (unsigned long long)(A + (size_t)(m0 + 64 + arow) * lda + ac * 8);
    unsigned long long Baddr =
        (unsigned long long)(W + (size_t)beff * ldb + bc * 8);

    const v8f vzero = {0.f, 0.f, 0.f, 0.f, 0.f, 0.f, 0.f, 0.f};
    v8f acc[2][2];
    acc[0][0] = vzero; acc[0][1] = vzero; acc[1][0] = vzero; acc[1][1] = vzero;

    // prologue: tile 0 -> buffer 0
    async_ld_b128(a0l, Aaddr0);
    async_ld_b128(a1l, Aaddr1);
    async_ld_b128(bl, Baddr);

    int cur = 0;
    for (int k0 = 0; k0 < K; k0 += TK) {
        const int nxt = cur ^ 1;
        if (k0 + TK < K) {
            Aaddr0 += TK * 2;
            Aaddr1 += TK * 2;
            Baddr += TK * 2;
            async_ld_b128(a0l + nxt * ASZ, Aaddr0);
            async_ld_b128(a1l + nxt * ASZ, Aaddr1);
            async_ld_b128(bl + nxt * BSZ, Baddr);
            if (k0 + 2 * TK < K)
                __builtin_prefetch(
                    A + (size_t)(m0 + (t >> 1)) * lda + k0 + 2 * TK, 0, 2);
            wait_async3(); // tile k complete (in-order), tile k+1 in flight
        } else {
            wait_async0();
        }
        __syncthreads();

        const unsigned short* Asc = As + cur * (TM * LDSP);
        const unsigned short* Bsc = Bs + cur * (TN * LDSP);

        // gather fragments per the CDNA5 16-bit WMMA VGPR layouts
        FragU au[2], bu[2];
#pragma unroll
        for (int mi = 0; mi < 2; ++mi) {
            int ar = wm * 32 + mi * 16 + l16;
#pragma unroll
            for (int p = 0; p < 8; ++p) {
                int grp = p >> 2;
                int off = (p & 3) * 2;
                int kk = grp * 16 + khalf * 8 + off;
                au[mi].u[p] = *(const unsigned int*)&Asc[ar * LDSP + kk];
            }
        }
#pragma unroll
        for (int ni = 0; ni < 2; ++ni) {
            int br = wn * 32 + ni * 16 + l16;
#pragma unroll
            for (int p = 0; p < 8; ++p) {
                int kk = (p >> 2) * 16 + khalf * 8 + (p & 3) * 2;
                bu[ni].u[p] = *(const unsigned int*)&Bsc[br * LDSP + kk];
            }
        }
#pragma unroll
        for (int mi = 0; mi < 2; ++mi)
#pragma unroll
            for (int ni = 0; ni < 2; ++ni)
                acc[mi][ni] = __builtin_amdgcn_wmma_f32_16x16x32_bf16(
                    false, au[mi].v, false, bu[ni].v, (short)0, acc[mi][ni],
                    false, false);
        __syncthreads(); // reads of buffer `cur` done before its reuse
        cur = nxt;
    }

    // epilogue: C/D layout -> lane holds col (lane&15), rows v + 8*(lane>>4)
#pragma unroll
    for (int mi = 0; mi < 2; ++mi) {
#pragma unroll
        for (int ni = 0; ni < 2; ++ni) {
            int cbase = n0 + wn * 32 + ni * 16 + l16;
            int rbase = m0 + wm * 32 + mi * 16 + khalf * 8;
#pragma unroll
            for (int v = 0; v < 8; ++v) {
                int r = rbase + v;
                int c = cbase;
                if (r < M && c < N) {
                    float x = acc[mi][ni][v];
                    if (flags & 1) x += bias[c];
                    if (flags & 2) x = 0.5f * x * (1.0f + erff(x * 0.70710678f));
                    if (flags & 4) x = (x > 20.0f) ? x : log1pf(__expf(x));
                    if (flags & 8) x += resid[(size_t)r * ldc + c];
                    if (flags & 16)
                        ((unsigned short*)Cout)[(size_t)r * ldc + c] = f2bf(x);
                    else
                        ((float*)Cout)[(size_t)r * ldc + c] = x;
                }
            }
        }
    }
}

// ---------------------------------------------------------------------------
// LayerNorm over last dim (D=1024) -> bf16 output
// ---------------------------------------------------------------------------
__global__ __launch_bounds__(256) void ln_bf16_kernel(
    const float* __restrict__ x, const float* __restrict__ g,
    const float* __restrict__ b, unsigned short* __restrict__ out, int D) {
    __shared__ float s1[256];
    __shared__ float s2[256];
    const int row = blockIdx.x;
    const float* xr = x + (size_t)row * D;
    float sum = 0.f, sq = 0.f;
    for (int i = threadIdx.x; i < D; i += 256) {
        float v = xr[i];
        sum += v;
        sq += v * v;
    }
    s1[threadIdx.x] = sum;
    s2[threadIdx.x] = sq;
    __syncthreads();
    for (int st = 128; st > 0; st >>= 1) {
        if (threadIdx.x < st) {
            s1[threadIdx.x] += s1[threadIdx.x + st];
            s2[threadIdx.x] += s2[threadIdx.x + st];
        }
        __syncthreads();
    }
    float mean = s1[0] / (float)D;
    float var = s2[0] / (float)D - mean * mean;
    float rstd = rsqrtf(var + 1e-5f);
    for (int i = threadIdx.x; i < D; i += 256) {
        float v = (xr[i] - mean) * rstd * g[i] + b[i];
        out[(size_t)row * D + i] = f2bf(v);
    }
}

// ---------------------------------------------------------------------------
// Causal depthwise conv (D_CONV=4) over L + bias + SiLU, reading x-half of xz
// ---------------------------------------------------------------------------
__global__ __launch_bounds__(256) void conv_silu_kernel(
    const float* __restrict__ xz, const float* __restrict__ cw,
    const float* __restrict__ cb, float* __restrict__ xcf,
    unsigned short* __restrict__ xcb) {
    int idx = blockIdx.x * 256 + threadIdx.x; // over B*L*DINNER
    int d = idx & (DINNER - 1);
    int bl = idx >> 11;
    int l = bl & (SEQL - 1);
    float s = cb[d];
#pragma unroll
    for (int k = 0; k < 4; ++k) {
        int ls = l - 3 + k;
        if (ls >= 0)
            s += cw[d * 4 + k] * xz[(size_t)(bl - 3 + k) * (2 * DINNER) + d];
    }
    s = silu_f(s);
    xcf[idx] = s;
    xcb[idx] = f2bf(s);
}

// ---------------------------------------------------------------------------
// A = -exp(A_log)
// ---------------------------------------------------------------------------
__global__ void aneg_kernel(const float* __restrict__ alog,
                            float* __restrict__ aneg, int n) {
    int i = blockIdx.x * 256 + threadIdx.x;
    if (i < n) aneg[i] = -__expf(alog[i]);
}

// ---------------------------------------------------------------------------
// f32 -> bf16 conversion
// ---------------------------------------------------------------------------
__global__ void cvt_bf16_kernel(const float* __restrict__ src,
                                unsigned short* __restrict__ dst, int n) {
    int i = blockIdx.x * 256 + threadIdx.x;
    if (i < n) dst[i] = f2bf(src[i]);
}

// ---------------------------------------------------------------------------
// Selective scan: one thread per (b, d). 16 states in registers, sequential L.
// Fuses D*x skip and SiLU(z) gating; emits bf16 for out_proj GEMM.
// ---------------------------------------------------------------------------
__global__ __launch_bounds__(256) void scan_kernel(
    const float* __restrict__ dt, const float* __restrict__ xc,
    const float* __restrict__ xz, const float* __restrict__ dbc,
    const float* __restrict__ aneg, const float* __restrict__ Dp,
    unsigned short* __restrict__ ybf) {
    const int gid = blockIdx.x * 256 + threadIdx.x; // 0..4095
    const int d = gid & (DINNER - 1);
    const int b = gid >> 11;
    __shared__ float sBC[32];
    float Areg[DSTATE];
#pragma unroll
    for (int n = 0; n < DSTATE; ++n) Areg[n] = aneg[d * DSTATE + n];
    const float Dd = Dp[d];
    float h[DSTATE];
#pragma unroll
    for (int n = 0; n < DSTATE; ++n) h[n] = 0.f;

    for (int l = 0; l < SEQL; ++l) {
        const size_t base = (size_t)(b * SEQL + l);
        if (threadIdx.x < 32)
            sBC[threadIdx.x] =
                dbc[base * (DTRANK + 2 * DSTATE) + DTRANK + threadIdx.x];
        __syncthreads();
        float dtv = dt[base * DINNER + d];
        float xv = xc[base * DINNER + d];
        float zv = xz[base * (2 * DINNER) + DINNER + d];
        float dtx = dtv * xv;
        float acc = 0.f;
#pragma unroll
        for (int n = 0; n < DSTATE; ++n) {
            float dA = __expf(dtv * Areg[n]);
            h[n] = dA * h[n] + dtx * sBC[n];
            acc += h[n] * sBC[DSTATE + n];
        }
        float y = acc + Dd * xv;
        ybf[base * DINNER + d] = f2bf(y * silu_f(zv));
        __syncthreads();
    }
}

// ---------------------------------------------------------------------------
// Host launcher
// ---------------------------------------------------------------------------
extern "C" void kernel_launch(void* const* d_in, const int* in_sizes, int n_in,
                              void* d_out, int out_size, void* d_ws,
                              size_t ws_size, hipStream_t stream) {
    const float* x = (const float*)d_in[0];
    const float* n1g = (const float*)d_in[1];
    const float* n1b = (const float*)d_in[2];
    const float* w_in = (const float*)d_in[3];
    const float* conv_w = (const float*)d_in[4];
    const float* conv_b = (const float*)d_in[5];
    const float* w_xp = (const float*)d_in[6];
    const float* w_dt = (const float*)d_in[7];
    const float* dt_b = (const float*)d_in[8];
    const float* a_log = (const float*)d_in[9];
    const float* d_par = (const float*)d_in[10];
    const float* w_out = (const float*)d_in[11];
    const float* n2g = (const float*)d_in[12];
    const float* n2b = (const float*)d_in[13];
    const float* w_m1 = (const float*)d_in[14];
    const float* b_m1 = (const float*)d_in[15];
    const float* w_m2 = (const float*)d_in[16];
    const float* b_m2 = (const float*)d_in[17];
    float* out = (float*)d_out;

    // workspace carve-up (256B aligned)
    size_t off = 0;
    char* base = (char*)d_ws;
    auto carve = [&](size_t bytes) -> void* {
        void* p = base + off;
        off = (off + bytes + 255) & ~(size_t)255;
        return p;
    };
    unsigned short* w_in_bf  = (unsigned short*)carve((size_t)2 * DINNER * DMODEL * 2);
    unsigned short* w_xp_bf  = (unsigned short*)carve((size_t)(DTRANK + 2 * DSTATE) * DINNER * 2);
    unsigned short* w_dt_bf  = (unsigned short*)carve((size_t)DINNER * DTRANK * 2);
    unsigned short* w_out_bf = (unsigned short*)carve((size_t)DMODEL * DINNER * 2);
    unsigned short* w_m1_bf  = (unsigned short*)carve((size_t)2 * DMODEL * DMODEL * 2);
    unsigned short* w_m2_bf  = (unsigned short*)carve((size_t)DMODEL * 2 * DMODEL * 2);
    float* aneg              = (float*)carve((size_t)DINNER * DSTATE * 4);
    unsigned short* xn_bf    = (unsigned short*)carve((size_t)MTOK * DMODEL * 2);
    float* xz                = (float*)carve((size_t)MTOK * 2 * DINNER * 4);
    float* xc_f              = (float*)carve((size_t)MTOK * DINNER * 4);
    unsigned short* xc_bf    = (unsigned short*)carve((size_t)MTOK * DINNER * 2); // reused as mlp1 out
    float* dbc               = (float*)carve((size_t)MTOK * (DTRANK + 2 * DSTATE) * 4);
    unsigned short* dbc_bf   = (unsigned short*)carve((size_t)MTOK * (DTRANK + 2 * DSTATE) * 2);
    float* dtbuf             = (float*)carve((size_t)MTOK * DINNER * 4);
    unsigned short* ybf      = (unsigned short*)carve((size_t)MTOK * DINNER * 2);
    float* hbuf              = (float*)carve((size_t)MTOK * DMODEL * 4);
    unsigned short* hn_bf    = (unsigned short*)carve((size_t)MTOK * DMODEL * 2);
    (void)ws_size; (void)n_in; (void)in_sizes; (void)out_size;

    // weight conversions to bf16
    auto cvt = [&](const float* s, unsigned short* d, int n) {
        cvt_bf16_kernel<<<(n + 255) / 256, 256, 0, stream>>>(s, d, n);
    };
    cvt(w_in, w_in_bf, 2 * DINNER * DMODEL);
    cvt(w_xp, w_xp_bf, (DTRANK + 2 * DSTATE) * DINNER);
    cvt(w_dt, w_dt_bf, DINNER * DTRANK);
    cvt(w_out, w_out_bf, DMODEL * DINNER);
    cvt(w_m1, w_m1_bf, 2 * DMODEL * DMODEL);
    cvt(w_m2, w_m2_bf, DMODEL * 2 * DMODEL);
    aneg_kernel<<<(DINNER * DSTATE + 255) / 256, 256, 0, stream>>>(
        a_log, aneg, DINNER * DSTATE);

    // 1. LN1
    ln_bf16_kernel<<<MTOK, 256, 0, stream>>>(x, n1g, n1b, xn_bf, DMODEL);

    // 2. xz = xn @ in_proj^T   (2048 x 4096 x 1024)
    {
        dim3 g(MTOK / TM, (2 * DINNER + TN - 1) / TN);
        gemm_bf16_wmma<<<g, 256, 0, stream>>>(xn_bf, w_in_bf, xz, nullptr,
                                              nullptr, MTOK, 2 * DINNER, DMODEL,
                                              DMODEL, DMODEL, 2 * DINNER, 0);
    }
    // 3. causal depthwise conv + SiLU
    conv_silu_kernel<<<(MTOK * DINNER) / 256, 256, 0, stream>>>(
        xz, conv_w, conv_b, xc_f, xc_bf);

    // 4. dbc = xc @ x_proj^T   (2048 x 96 x 2048)
    {
        dim3 g(MTOK / TM, (DTRANK + 2 * DSTATE + TN - 1) / TN);
        gemm_bf16_wmma<<<g, 256, 0, stream>>>(
            xc_bf, w_xp_bf, dbc, nullptr, nullptr, MTOK, DTRANK + 2 * DSTATE,
            DINNER, DINNER, DINNER, DTRANK + 2 * DSTATE, 0);
    }
    cvt(dbc, dbc_bf, MTOK * (DTRANK + 2 * DSTATE));

    // 5. dt = softplus(dbc[:, :64] @ dt_proj^T + dt_b)  (2048 x 2048 x 64)
    {
        dim3 g(MTOK / TM, (DINNER + TN - 1) / TN);
        gemm_bf16_wmma<<<g, 256, 0, stream>>>(
            dbc_bf, w_dt_bf, dtbuf, dt_b, nullptr, MTOK, DINNER, DTRANK,
            DTRANK + 2 * DSTATE, DTRANK, DINNER, 1 | 4);
    }
    // 6. selective scan (fused D*x skip + SiLU(z) gate) -> bf16
    scan_kernel<<<(BATCH * DINNER) / 256, 256, 0, stream>>>(
        dtbuf, xc_f, xz, dbc, aneg, d_par, ybf);

    // 7. h = x + y @ out_proj^T   (2048 x 1024 x 2048)
    {
        dim3 g(MTOK / TM, (DMODEL + TN - 1) / TN);
        gemm_bf16_wmma<<<g, 256, 0, stream>>>(ybf, w_out_bf, hbuf, nullptr, x,
                                              MTOK, DMODEL, DINNER, DINNER,
                                              DINNER, DMODEL, 8);
    }
    // 8. LN2
    ln_bf16_kernel<<<MTOK, 256, 0, stream>>>(hbuf, n2g, n2b, hn_bf, DMODEL);

    // 9. g = gelu(hn @ mlp_w1^T + b1) -> bf16 (reuse xc_bf)
    {
        dim3 g(MTOK / TM, (2 * DMODEL + TN - 1) / TN);
        gemm_bf16_wmma<<<g, 256, 0, stream>>>(hn_bf, w_m1_bf, xc_bf, b_m1,
                                              nullptr, MTOK, 2 * DMODEL, DMODEL,
                                              DMODEL, DMODEL, 2 * DMODEL,
                                              1 | 2 | 16);
    }
    // 10. out = h + g @ mlp_w2^T + b2   (2048 x 1024 x 2048)
    {
        dim3 g(MTOK / TM, (DMODEL + TN - 1) / TN);
        gemm_bf16_wmma<<<g, 256, 0, stream>>>(xc_bf, w_m2_bf, out, b_m2, hbuf,
                                              MTOK, DMODEL, 2 * DMODEL,
                                              2 * DMODEL, 2 * DMODEL, DMODEL,
                                              1 | 8);
    }
}
